// Retriever_26731876450714
// MI455X (gfx1250) — compile-verified, matrix operand order checked by reading, not run on hardware
//
#include <hip/hip_runtime.h>
#include <math.h>

typedef __bf16 bf16;
typedef __attribute__((ext_vector_type(16))) __bf16 v16bf;
typedef __attribute__((ext_vector_type(8)))  __bf16 v8bf;
typedef __attribute__((ext_vector_type(8)))  float  v8f;

#define EMB 768
#define SRAW 20
#define WAVES 8
#define NTW 6          // N-tiles per wave: 6*16 = 96 columns
#define LDS_STR 40     // LDS tile row stride (bf16 elems) -> conflict-free b128 frag loads

static __device__ __forceinline__ float bf2f(bf16 x) { return (float)x; }
static __device__ __forceinline__ bf16  f2bf(float x) { return (bf16)x; }
static __device__ __forceinline__ float gelu_exact(float x) {
    return 0.5f * x * (1.0f + erff(x * 0.70710678118654752f));
}
static __device__ __forceinline__ float sigmoidf_(float x) {
    return 1.0f / (1.0f + __expf(-x));
}

// ---------------------------------------------------------------------------
// CDNA5 async global->LDS copy (GLOBAL_LOAD_ASYNC_TO_LDS_B128, ASYNCcnt) via
// inline asm (portable across ROCm/upstream toolchains per the bridge doc).
// lds_off is the 32-bit LDS byte offset (low 32 bits of the shared-aperture
// flat address); g is the per-lane 64-bit global address (GV mode, saddr=off).
// ---------------------------------------------------------------------------
static __device__ __forceinline__ void async_ld_b128(unsigned lds_off, const void* g) {
    asm volatile("global_load_async_to_lds_b128 %0, %1, off"
                 :: "v"(lds_off), "v"(g) : "memory");
}
static __device__ __forceinline__ void wait_async0() {
    asm volatile("s_wait_asynccnt 0" ::: "memory");
}

// ---------------------------------------------------------------------------
// Pack a row-major f32 B[K,768] into WMMA-fragment-major bf16:
// fragment f = ktile*48 + ntile holds 32 lanes x 16 elems contiguous, so each
// lane reads its whole B fragment as one 32-byte contiguous load.
// ---------------------------------------------------------------------------
__global__ void pack_b_kernel(const float* __restrict__ B, bf16* __restrict__ pk, int K) {
    int tid = blockIdx.x * blockDim.x + threadIdx.x;
    if (tid >= K * EMB) return;
    int f = tid >> 9;      // 512 elems / fragment
    int r = tid & 511;
    int l = r >> 4;        // lane 0..31
    int i = r & 15;        // elem 0..15
    int kt = f / 48;
    int nt = f % 48;
    int koff = ((l >> 4) << 3) + ((i >> 3) << 4) + (i & 7);
    int k = kt * 32 + koff;
    int n = nt * 16 + (l & 15);
    pk[tid] = f2bf(B[(size_t)k * EMB + n]);
}

// ---------------------------------------------------------------------------
// q_all = question @ qry_w + qry_b   (16 rows -> trivial VALU kernel)
// ---------------------------------------------------------------------------
__global__ void qproj_kernel(const float* __restrict__ x, const float* __restrict__ W,
                             const float* __restrict__ bias, float* __restrict__ out) {
    __shared__ float xs[EMB];
    int b = blockIdx.x, tid = threadIdx.x;
    for (int i = tid; i < EMB; i += 256) xs[i] = x[(size_t)b * EMB + i];
    __syncthreads();
    for (int n = tid; n < EMB; n += 256) {
        float acc = bias[n];
        for (int k = 0; k < EMB; ++k) acc += xs[k] * W[(size_t)k * EMB + n];
        out[(size_t)b * EMB + n] = acc;
    }
}

// ---------------------------------------------------------------------------
// gate_all = sigmoid(q_all @ qg_w + qg_b); bias_all = tanh(q_all @ qb_w + qb_b)
// (hoisted to the 16 query rows instead of 65536 edges)
// ---------------------------------------------------------------------------
__global__ void gatebias_kernel(const float* __restrict__ q,
                                const float* __restrict__ gw, const float* __restrict__ gb,
                                const float* __restrict__ bw, const float* __restrict__ bb,
                                float* __restrict__ gate, float* __restrict__ biaso) {
    __shared__ float xs[EMB];
    int b = blockIdx.x, tid = threadIdx.x;
    for (int i = tid; i < EMB; i += 256) xs[i] = q[(size_t)b * EMB + i];
    __syncthreads();
    for (int n = tid; n < EMB; n += 256) {
        float a = gb[n], c = bb[n];
        for (int k = 0; k < EMB; ++k) {
            float xv = xs[k];
            a += xv * gw[(size_t)k * EMB + n];
            c += xv * bw[(size_t)k * EMB + n];
        }
        gate[(size_t)b * EMB + n]  = sigmoidf_(a);
        biaso[(size_t)b * EMB + n] = tanhf(c);
    }
}

// ---------------------------------------------------------------------------
// struct branch: s = gelu(LN(raw @ sp_w + sp_b)); sg = sigmoid(s @ sg_w + sg_b)
// one 256-thread block per edge; K=20 so VALU is the right tool.
// ---------------------------------------------------------------------------
__global__ void struct_kernel(const float* __restrict__ raw,
                              const float* __restrict__ spw, const float* __restrict__ spb,
                              const float* __restrict__ g, const float* __restrict__ be,
                              const float* __restrict__ sgw, const float* __restrict__ sgb,
                              bf16* __restrict__ sOut, float* __restrict__ sgOut) {
    __shared__ float rs[SRAW];
    __shared__ float red[3];
    int e = blockIdx.x, tid = threadIdx.x;
    if (tid < SRAW) rs[tid] = raw[(size_t)e * SRAW + tid];
    if (tid < 3) red[tid] = 0.f;
    __syncthreads();
    float v[3];
    float ps = 0.f, pq = 0.f;
#pragma unroll
    for (int j = 0; j < 3; ++j) {
        int n = tid + j * 256;
        float a = spb[n];
#pragma unroll
        for (int k = 0; k < SRAW; ++k) a += rs[k] * spw[k * EMB + n];
        v[j] = a; ps += a; pq += a * a;
    }
    for (int m = 1; m < 32; m <<= 1) { ps += __shfl_xor(ps, m, 32); pq += __shfl_xor(pq, m, 32); }
    if ((tid & 31) == 0) { atomicAdd(&red[0], ps); atomicAdd(&red[1], pq); }
    __syncthreads();
    float mu = red[0] * (1.f / EMB);
    float var = red[1] * (1.f / EMB) - mu * mu;
    float rstd = rsqrtf(var + 1e-5f);
    float pd = 0.f;
#pragma unroll
    for (int j = 0; j < 3; ++j) {
        int n = tid + j * 256;
        float s = gelu_exact((v[j] - mu) * rstd * g[n] + be[n]);
        sOut[(size_t)e * EMB + n] = f2bf(s);
        pd += s * sgw[n];
    }
    for (int m = 1; m < 32; m <<= 1) pd += __shfl_xor(pd, m, 32);
    if ((tid & 31) == 0) atomicAdd(&red[2], pd);
    __syncthreads();
    if (tid == 0) sgOut[e] = sigmoidf_(red[2] + sgb[0]);
}

// ---------------------------------------------------------------------------
// Generic bf16 WMMA GEMM: C_bf16[M,768] = A[M,K] @ packedB[K,768] + bias.
// A f32: converted while staging in LDS (manual path).
// A bf16: staged global->LDS with GLOBAL_LOAD_ASYNC_TO_LDS_B128 (ASYNCcnt).
// Block = 256 thr = 8 waves, M_TILE = 16, each wave covers 96 columns.
// ---------------------------------------------------------------------------
template <bool AF32>
__global__ __launch_bounds__(256)
void gemm768_kernel(const void* __restrict__ Ain, const bf16* __restrict__ pkB,
                    const float* __restrict__ bias, bf16* __restrict__ C,
                    int M, int K) {
    __shared__ __align__(16) bf16 tA[16][LDS_STR];
    const int tid = threadIdx.x;
    const int lane = tid & 31, wave = tid >> 5;
    const int mrow = tid >> 4;
    const int ccol = (tid & 15) * 2;
    const int r0 = blockIdx.x * 16;
    int arow = r0 + mrow; if (arow >= M) arow = M - 1;
    const float* Af = (const float*)Ain;
    const bf16*  Ab = (const bf16*)Ain;

    // async-staging assignment: 64 lanes move the whole 16x32 bf16 tile as b128s
    int am = tid >> 2, aq = tid & 3;
    int aarow = r0 + am; if (aarow >= M) aarow = M - 1;
    unsigned aLds = (unsigned)(size_t)(&tA[0][0]) + (unsigned)(am * (LDS_STR * 2) + aq * 16);

    v8f acc[NTW];
#pragma unroll
    for (int nt = 0; nt < NTW; ++nt)
#pragma unroll
        for (int j = 0; j < 8; ++j) acc[nt][j] = 0.f;

    const int chunks = K >> 5;
    for (int cc = 0; cc < chunks; ++cc) {
        int kk = cc * 32;
        if (AF32) {
            float2 d = *(const float2*)(Af + (size_t)arow * K + kk + ccol);
            tA[mrow][ccol] = f2bf(d.x); tA[mrow][ccol + 1] = f2bf(d.y);
            if (cc + 1 < chunks)
                __builtin_prefetch((const char*)(Af + (size_t)arow * K + kk + 32 + ccol), 0, 1);
        } else {
            if (tid < 64)
                async_ld_b128(aLds, Ab + (size_t)aarow * K + kk + aq * 8);
            wait_async0();
        }
        __syncthreads();
        const int mfr = lane & 15, kh = (lane >> 4) * 8;
        v8bf lo = *(const v8bf*)&tA[mfr][kh];
        v8bf hi = *(const v8bf*)&tA[mfr][kh + 16];
        v16bf afrag;
#pragma unroll
        for (int i = 0; i < 8; ++i) { afrag[i] = lo[i]; afrag[i + 8] = hi[i]; }
#pragma unroll
        for (int nt = 0; nt < NTW; ++nt) {
            size_t f = (size_t)cc * 48 + (wave * NTW + nt);
            v16bf bfrag = *(const v16bf*)(pkB + f * 512 + lane * 16);
            acc[nt] = __builtin_amdgcn_wmma_f32_16x16x32_bf16(
                false, afrag, false, bfrag, (short)0, acc[nt], false, false);
        }
        __syncthreads();
    }
#pragma unroll
    for (int nt = 0; nt < NTW; ++nt) {
        int n = wave * 96 + nt * 16 + (lane & 15);
        float bv = bias[n];
#pragma unroll
        for (int j = 0; j < 8; ++j) {
            int m = j + ((lane >> 4) << 3);
            int row = r0 + m;
            if (row < M) C[(size_t)row * EMB + n] = f2bf(acc[nt][j] + bv);
        }
    }
}

// ---------------------------------------------------------------------------
// Fused state_net layer-1 for one direction:
//   A tiles (base/dm/diff) assembled on the fly from gathered h/t/rel/s +
//   hoisted gate/bias, K-split WMMA against packed s1_w, transe rank-1 update
//   + bias + row LayerNorm + GELU in the epilogue.  hdn out in bf16.
// ---------------------------------------------------------------------------
__global__ __launch_bounds__(256)
void fused_state1_kernel(const bf16* __restrict__ nodeR, const bf16* __restrict__ relR,
                         const bf16* __restrict__ sB, const float* __restrict__ sg,
                         const float* __restrict__ gateAll, const float* __restrict__ biasAll,
                         const int* __restrict__ hIdxG, const int* __restrict__ tIdxG,
                         const int* __restrict__ qIdG,
                         const bf16* __restrict__ pkS1, const float* __restrict__ s1w,
                         const float* __restrict__ s1b, const float* __restrict__ s1g,
                         const float* __restrict__ s1be,
                         bf16* __restrict__ hdn, int E) {
    __shared__ __align__(16) bf16 tBase[16][LDS_STR];
    __shared__ __align__(16) bf16 tDm[16][LDS_STR];
    __shared__ __align__(16) bf16 tDf[16][LDS_STR];
    __shared__ int hI[16], tI[16], qI[16];
    __shared__ float sgv[16], ssq[16], tre[16], rSum[16], rSsq[16];

    const int tid = threadIdx.x;
    const int lane = tid & 31, wave = tid >> 5;
    const int mrow = tid >> 4, cc2 = (tid & 15) * 2;
    const int e0 = blockIdx.x * 16;

    if (tid < 16) {
        int e = e0 + tid; if (e >= E) e = E - 1;
        hI[tid] = hIdxG[e]; tI[tid] = tIdxG[e]; qI[tid] = qIdG[e];
        sgv[tid] = sg[e]; ssq[tid] = 0.f;
    }
    __syncthreads();

    int e = e0 + mrow; if (e >= E) e = E - 1;
    const bf16*  relRow = relR + (size_t)e * EMB;
    const bf16*  sRow   = sB   + (size_t)e * EMB;
    const bf16*  hRow   = nodeR + (size_t)hI[mrow] * EMB;
    const bf16*  tRow   = nodeR + (size_t)tI[mrow] * EMB;
    const float* gRow   = gateAll + (size_t)qI[mrow] * EMB;
    const float* bRow   = biasAll + (size_t)qI[mrow] * EMB;
    const float  sgm    = sgv[mrow];
    float myssq = 0.f;

    v8f acc[NTW];
#pragma unroll
    for (int nt = 0; nt < NTW; ++nt)
#pragma unroll
        for (int j = 0; j < 8; ++j) acc[nt][j] = 0.f;

    for (int cc = 0; cc < 24; ++cc) {
        const int kk = cc * 32;
#pragma unroll
        for (int u = 0; u < 2; ++u) {
            int c = cc2 + u;
            float r  = bf2f(relRow[kk + c]);
            float h  = bf2f(hRow[kk + c]);
            float t  = bf2f(tRow[kk + c]);
            float s  = bf2f(sRow[kk + c]);
            float rel = gRow[kk + c] * r + bRow[kk + c];
            float bas = rel + sgm * s;
            float dm  = h * rel * t;
            float df  = h + rel - t;
            myssq += df * df;
            tBase[mrow][c] = f2bf(bas);
            tDm[mrow][c]   = f2bf(dm);
            tDf[mrow][c]   = f2bf(df);
        }
        __syncthreads();
        const int mfr = lane & 15, kh = (lane >> 4) * 8;
        v16bf aB, aD, aF;
        { v8bf lo = *(const v8bf*)&tBase[mfr][kh]; v8bf hi = *(const v8bf*)&tBase[mfr][kh + 16];
#pragma unroll
          for (int i = 0; i < 8; ++i) { aB[i] = lo[i]; aB[i + 8] = hi[i]; } }
        { v8bf lo = *(const v8bf*)&tDm[mfr][kh];   v8bf hi = *(const v8bf*)&tDm[mfr][kh + 16];
#pragma unroll
          for (int i = 0; i < 8; ++i) { aD[i] = lo[i]; aD[i + 8] = hi[i]; } }
        { v8bf lo = *(const v8bf*)&tDf[mfr][kh];   v8bf hi = *(const v8bf*)&tDf[mfr][kh + 16];
#pragma unroll
          for (int i = 0; i < 8; ++i) { aF[i] = lo[i]; aF[i + 8] = hi[i]; } }
#pragma unroll
        for (int nt = 0; nt < NTW; ++nt) {
            const int ntg = wave * NTW + nt;
            size_t f0 = ((size_t)(cc)) * 48 + ntg;        // base  segment ktile = cc
            size_t f1 = ((size_t)(24 + cc)) * 48 + ntg;   // dm    segment
            size_t f2 = ((size_t)(48 + cc)) * 48 + ntg;   // diff  segment
            v16bf b0 = *(const v16bf*)(pkS1 + f0 * 512 + lane * 16);
            acc[nt] = __builtin_amdgcn_wmma_f32_16x16x32_bf16(false, aB, false, b0, (short)0, acc[nt], false, false);
            v16bf b1 = *(const v16bf*)(pkS1 + f1 * 512 + lane * 16);
            acc[nt] = __builtin_amdgcn_wmma_f32_16x16x32_bf16(false, aD, false, b1, (short)0, acc[nt], false, false);
            v16bf b2 = *(const v16bf*)(pkS1 + f2 * 512 + lane * 16);
            acc[nt] = __builtin_amdgcn_wmma_f32_16x16x32_bf16(false, aF, false, b2, (short)0, acc[nt], false, false);
        }
        __syncthreads();
    }

    // transe = -sqrt(sum diff^2) per row
    atomicAdd(&ssq[mrow], myssq);
    __syncthreads();
    if (tid < 16) { tre[tid] = -sqrtf(ssq[tid]); rSum[tid] = 0.f; rSsq[tid] = 0.f; }
    __syncthreads();

    // epilogue: rank-1 transe term + bias, then cross-wave LayerNorm partials
    float val[NTW][8];
    float psum[8], pssq[8];
#pragma unroll
    for (int j = 0; j < 8; ++j) { psum[j] = 0.f; pssq[j] = 0.f; }
#pragma unroll
    for (int nt = 0; nt < NTW; ++nt) {
        int n = wave * 96 + nt * 16 + (lane & 15);
        float wl = s1w[(size_t)2304 * EMB + n];
        float bb = s1b[n];
#pragma unroll
        for (int j = 0; j < 8; ++j) {
            int m = j + ((lane >> 4) << 3);
            float v = acc[nt][j] + tre[m] * wl + bb;
            val[nt][j] = v; psum[j] += v; pssq[j] += v * v;
        }
    }
#pragma unroll
    for (int j = 0; j < 8; ++j)
        for (int msk = 1; msk < 16; msk <<= 1) {
            psum[j] += __shfl_xor(psum[j], msk, 32);
            pssq[j] += __shfl_xor(pssq[j], msk, 32);
        }
    if ((lane & 15) == 0) {
        int mb = (lane >> 4) << 3;
#pragma unroll
        for (int j = 0; j < 8; ++j) { atomicAdd(&rSum[mb + j], psum[j]); atomicAdd(&rSsq[mb + j], pssq[j]); }
    }
    __syncthreads();
#pragma unroll
    for (int nt = 0; nt < NTW; ++nt) {
        int n = wave * 96 + nt * 16 + (lane & 15);
        float gn = s1g[n], bn = s1be[n];
#pragma unroll
        for (int j = 0; j < 8; ++j) {
            int m = j + ((lane >> 4) << 3);
            int row = e0 + m;
            if (row < E) {
                float mu  = rSum[m] * (1.f / EMB);
                float var = rSsq[m] * (1.f / EMB) - mu * mu;
                float rstd = rsqrtf(var + 1e-5f);
                float x = (val[nt][j] - mu) * rstd * gn + bn;
                hdn[(size_t)row * EMB + n] = f2bf(gelu_exact(x));
            }
        }
    }
}

// ---------------------------------------------------------------------------
// score head + fwd/bwd softmax combine + edge_emb.
// d_out layout: [logits E | logits_fwd E | logits_bwd E | edge_emb E*768] f32
// ---------------------------------------------------------------------------
__global__ void combine_kernel(const bf16* __restrict__ stF, const bf16* __restrict__ stB,
                               const float* __restrict__ shw, const float* __restrict__ shb,
                               float* __restrict__ out, int E) {
    __shared__ float red[2];
    __shared__ float wgt[2];
    int e = blockIdx.x, tid = threadIdx.x;
    if (tid < 2) red[tid] = 0.f;
    __syncthreads();
    const bf16* fr = stF + (size_t)e * EMB;
    const bf16* br = stB + (size_t)e * EMB;
    float pf = 0.f, pb = 0.f;
#pragma unroll
    for (int j = 0; j < 3; ++j) {
        int n = tid + j * 256;
        float w = shw[n];
        pf += bf2f(fr[n]) * w; pb += bf2f(br[n]) * w;
    }
    for (int m = 1; m < 32; m <<= 1) { pf += __shfl_xor(pf, m, 32); pb += __shfl_xor(pb, m, 32); }
    if ((tid & 31) == 0) { atomicAdd(&red[0], pf); atomicAdd(&red[1], pb); }
    __syncthreads();
    float lf = red[0] + shb[0];
    float lb = red[1] + shb[0];
    if (tid == 0) {
        float mx = fmaxf(lf, lb);
        float e0 = __expf(lf - mx), e1 = __expf(lb - mx);
        float inv = 1.f / (e0 + e1);
        wgt[0] = e0 * inv; wgt[1] = e1 * inv;
        out[e] = wgt[0] * lf + wgt[1] * lb;
        out[(size_t)E + e] = lf;
        out[(size_t)2 * E + e] = lb;
    }
    __syncthreads();
    float w0 = wgt[0], w1 = wgt[1];
    float* eo = out + (size_t)3 * E + (size_t)e * EMB;
#pragma unroll
    for (int j = 0; j < 3; ++j) {
        int n = tid + j * 256;
        eo[n] = w0 * bf2f(fr[n]) + w1 * bf2f(br[n]);
    }
}

// ---------------------------------------------------------------------------
extern "C" void kernel_launch(void* const* d_in, const int* in_sizes, int n_in,
                              void* d_out, int out_size, void* d_ws, size_t ws_size,
                              hipStream_t stream) {
    const float* question   = (const float*)d_in[0];
    const float* node_emb   = (const float*)d_in[1];
    const float* edge_emb   = (const float*)d_in[2];
    const int*   head_idx   = (const int*)d_in[3];
    const int*   tail_idx   = (const int*)d_in[4];
    const int*   query_ids  = (const int*)d_in[5];
    const float* struct_fwd = (const float*)d_in[6];
    const float* struct_bwd = (const float*)d_in[7];
    const float* ent_w = (const float*)d_in[8];
    const float* ent_b = (const float*)d_in[9];
    const float* rel_w = (const float*)d_in[10];
    const float* rel_b = (const float*)d_in[11];
    const float* qry_w = (const float*)d_in[12];
    const float* qry_b = (const float*)d_in[13];
    const float* qg_w  = (const float*)d_in[14];
    const float* qg_b  = (const float*)d_in[15];
    const float* qb_w  = (const float*)d_in[16];
    const float* qb_b  = (const float*)d_in[17];
    const float* sp_w  = (const float*)d_in[18];
    const float* sp_b  = (const float*)d_in[19];
    const float* sp_g  = (const float*)d_in[20];
    const float* sp_be = (const float*)d_in[21];
    const float* sg_w  = (const float*)d_in[22];
    const float* sg_b  = (const float*)d_in[23];
    const float* s1_w  = (const float*)d_in[24];
    const float* s1_b  = (const float*)d_in[25];
    const float* s1_g  = (const float*)d_in[26];
    const float* s1_be = (const float*)d_in[27];
    const float* s2_w  = (const float*)d_in[28];
    const float* s2_b  = (const float*)d_in[29];
    const float* sh_w  = (const float*)d_in[30];
    const float* sh_b  = (const float*)d_in[31];

    const int Bq = in_sizes[0] / EMB;     // 16
    const int Nn = in_sizes[1] / EMB;     // 50000
    const int E  = in_sizes[3];           // 65536

    // ---- workspace carve (256B aligned) ----
    char* p = (char*)d_ws;
    auto carve = [&](size_t bytes) -> char* {
        char* r = p;
        p += (bytes + 255) & ~(size_t)255;
        return r;
    };
    float* q_all    = (float*)carve((size_t)Bq * EMB * 4);
    float* gate_all = (float*)carve((size_t)Bq * EMB * 4);
    float* bias_all = (float*)carve((size_t)Bq * EMB * 4);
    bf16* pk_ent = (bf16*)carve((size_t)EMB * EMB * 2);
    bf16* pk_rel = (bf16*)carve((size_t)EMB * EMB * 2);
    bf16* pk_s2  = (bf16*)carve((size_t)EMB * EMB * 2);
    bf16* pk_s1  = (bf16*)carve((size_t)2304 * EMB * 2);
    bf16* node_repr = (bf16*)carve((size_t)(((Nn + 15) & ~15)) * EMB * 2);
    bf16* rel_repr  = (bf16*)carve((size_t)E * EMB * 2);
    bf16* s_buf     = (bf16*)carve((size_t)E * EMB * 2);
    float* sg_buf   = (float*)carve((size_t)E * 4);
    bf16* hdn_buf   = (bf16*)carve((size_t)E * EMB * 2);
    bf16* state_fwd = (bf16*)carve((size_t)E * EMB * 2);
    bf16* state_bwd = (bf16*)carve((size_t)E * EMB * 2);
    (void)ws_size; (void)n_in; (void)out_size;

    // ---- 1. pack weight matrices into WMMA fragment-major bf16 ----
    pack_b_kernel<<<(EMB * EMB + 255) / 256, 256, 0, stream>>>(ent_w, pk_ent, EMB);
    pack_b_kernel<<<(EMB * EMB + 255) / 256, 256, 0, stream>>>(rel_w, pk_rel, EMB);
    pack_b_kernel<<<(EMB * EMB + 255) / 256, 256, 0, stream>>>(s2_w,  pk_s2,  EMB);
    pack_b_kernel<<<(2304 * EMB + 255) / 256, 256, 0, stream>>>(s1_w, pk_s1, 2304);

    // ---- 2. tiny query-side projections (hoisted off the edge dimension) ----
    qproj_kernel<<<Bq, 256, 0, stream>>>(question, qry_w, qry_b, q_all);
    gatebias_kernel<<<Bq, 256, 0, stream>>>(q_all, qg_w, qg_b, qb_w, qb_b, gate_all, bias_all);

    // ---- 3. node / relation projections (bf16 WMMA GEMMs, f32 A staging) ----
    gemm768_kernel<true><<<(Nn + 15) / 16, 256, 0, stream>>>(node_emb, pk_ent, ent_b, node_repr, Nn, EMB);
    gemm768_kernel<true><<<(E + 15) / 16, 256, 0, stream>>>(edge_emb, pk_rel, rel_b, rel_repr, E, EMB);

    // ---- 4. per-direction pipeline ----
    const int eblk = (E + 15) / 16;
    // forward: h = head, t = tail
    struct_kernel<<<E, 256, 0, stream>>>(struct_fwd, sp_w, sp_b, sp_g, sp_be, sg_w, sg_b, s_buf, sg_buf);
    fused_state1_kernel<<<eblk, 256, 0, stream>>>(node_repr, rel_repr, s_buf, sg_buf,
                                                  gate_all, bias_all, head_idx, tail_idx, query_ids,
                                                  pk_s1, s1_w, s1_b, s1_g, s1_be, hdn_buf, E);
    gemm768_kernel<false><<<eblk, 256, 0, stream>>>(hdn_buf, pk_s2, s2_b, state_fwd, E, EMB);
    // backward: h = tail, t = head
    struct_kernel<<<E, 256, 0, stream>>>(struct_bwd, sp_w, sp_b, sp_g, sp_be, sg_w, sg_b, s_buf, sg_buf);
    fused_state1_kernel<<<eblk, 256, 0, stream>>>(node_repr, rel_repr, s_buf, sg_buf,
                                                  gate_all, bias_all, tail_idx, head_idx, query_ids,
                                                  pk_s1, s1_w, s1_b, s1_g, s1_be, hdn_buf, E);
    gemm768_kernel<false><<<eblk, 256, 0, stream>>>(hdn_buf, pk_s2, s2_b, state_bwd, E, EMB);

    // ---- 5. score head + softmax combine + edge_emb ----
    combine_kernel<<<E, 256, 0, stream>>>(state_fwd, state_bwd, sh_w, sh_b, (float*)d_out, E);
}
